// HHGNN_17179869473
// MI455X (gfx1250) — matrix operand order repaired and checked by reading, not compiled, not run on hardware
//
#include <hip/hip_runtime.h>

#define HIDDEN 128
#define NTYPES 13
#define BATCH 8
#define NN 4096
#define MM 2048
#define LN_EPS 1e-5f

typedef __attribute__((ext_vector_type(16))) __bf16 v16bf;
typedef __attribute__((ext_vector_type(16))) unsigned short v16u;
typedef __attribute__((ext_vector_type(8))) float v8f;

struct U32x8 { uint4 lo; uint4 hi; };   // 32B container for an A/B operand

__device__ __forceinline__ unsigned short f2bf(float f) {
  unsigned int u = __float_as_uint(f);
  u += 0x7FFFu + ((u >> 16) & 1u);   // round-to-nearest-even
  return (unsigned short)(u >> 16);
}

// ---------------------------------------------------------------------------
// Stage 0: per-node typed linear + bias + mask + LayerNorm -> xT bf16 [B][H][N]
// One block (128 threads) per node; W column reads are coalesced across d.
// ---------------------------------------------------------------------------
__global__ void __launch_bounds__(HIDDEN)
stage0_linear_ln(const float* __restrict__ embeds,
                 const int* __restrict__ tindex,
                 const unsigned char* __restrict__ mask,
                 const float* __restrict__ W,
                 const float* __restrict__ bias,
                 const float* __restrict__ gamma,
                 const float* __restrict__ beta,
                 unsigned short* __restrict__ xT) {
  __shared__ float eS[HIDDEN];
  __shared__ float r1[HIDDEN];
  __shared__ float r2[HIDDEN];
  const int node = blockIdx.x;            // b*N + n
  const int b = node / NN;
  const int n = node - b * NN;
  const int d = threadIdx.x;

  const float e_d = embeds[(size_t)node * HIDDEN + d];
  eS[d] = e_d;
  __syncthreads();

  const int t = tindex[node];
  float acc = bias[t * HIDDEN + d];
  const float* wp = W + ((size_t)t * HIDDEN) * HIDDEN + d;
#pragma unroll 8
  for (int h = 0; h < HIDDEN; ++h)
    acc = fmaf(eS[h], wp[(size_t)h * HIDDEN], acc);

  const float val = mask[node] ? acc : e_d;

  r1[d] = val;
  r2[d] = val * val;
  __syncthreads();
#pragma unroll
  for (int s = HIDDEN / 2; s > 0; s >>= 1) {
    if (d < s) { r1[d] += r1[d + s]; r2[d] += r2[d + s]; }
    __syncthreads();
  }
  const float mu  = r1[0] * (1.0f / HIDDEN);
  const float var = r2[0] * (1.0f / HIDDEN) - mu * mu;
  const float rs  = rsqrtf(var + LN_EPS);
  const float xn  = (val - mu) * rs * gamma[d] + beta[d];

  xT[((size_t)b * HIDDEN + d) * NN + n] = f2bf(xn);
}

// ---------------------------------------------------------------------------
// Stage 1: latT[b][h][m] = sum_n adj[b][n][m] * x[b][n][h]
// LDS holds the adj tile TRANSPOSED as bf16 [m_local 32][n_local 128] so the
// WMMA A operand is two aligned ds_load_b128 per 16x16x32 step.
// WG = 256 thr: 32(m) x 128(h) output tile, K-loop over N in steps of 128.
// ---------------------------------------------------------------------------
#define PADK 136   // row pad (u16) -> 272B row stride, keeps 16B alignment

__global__ void __launch_bounds__(256)
stage1_aggT(const float* __restrict__ adj,
            const unsigned short* __restrict__ xT,
            unsigned short* __restrict__ latT) {
  __shared__ unsigned short adjS[32 * PADK];  // bf16, [m_local][n_local]
  __shared__ unsigned short latS[128 * 40];   // output transpose staging

  const int b   = blockIdx.y;
  const int m0  = blockIdx.x * 32;
  const int tid = threadIdx.x;
  const int wv = tid >> 5, lane = tid & 31;
  const int l15 = lane & 15, hi = lane >> 4;
  const int msub = wv & 1;            // 16-wide m sub-tile
  const int ht0  = (wv >> 1) * 2;     // pair of 16-wide h tiles
  const int mloc = msub * 16 + l15;

  // transposed-fill mapping: each thread covers m = {2*mduo, 2*mduo+1},
  // n-pairs np, np+16, np+32, np+48 (n = 2*np, 2*np+1)
  const int mduo = tid & 15;
  const int npb  = tid >> 4;          // 0..15
  const float* fb = adj + ((size_t)b * NN) * MM + m0 + 2 * mduo;

  v8f acc0 = {}, acc1 = {};

  for (int kk = 0; kk < NN / 128; ++kk) {
    const int n0 = kk * 128;
    // ---- fill LDS with bf16 TRANSPOSED tile ----
#pragma unroll
    for (int tt = 0; tt < 4; ++tt) {
      const int np = npb + 16 * tt;                    // 0..63
      const float2 r0 = *(const float2*)(fb + (size_t)(n0 + 2 * np)     * MM);
      const float2 r1 = *(const float2*)(fb + (size_t)(n0 + 2 * np + 1) * MM);
      const unsigned int d0 =
          (unsigned int)f2bf(r0.x) | ((unsigned int)f2bf(r1.x) << 16);
      const unsigned int d1 =
          (unsigned int)f2bf(r0.y) | ((unsigned int)f2bf(r1.y) << 16);
      *(unsigned int*)(&adjS[(2 * mduo)     * PADK + 2 * np]) = d0;
      *(unsigned int*)(&adjS[(2 * mduo + 1) * PADK + 2 * np]) = d1;
    }
    __builtin_prefetch(fb + (size_t)(n0 + 128 + 2 * npb) * MM, 0, 1);
    __syncthreads();

#pragma unroll
    for (int kh = 0; kh < 4; ++kh) {
      // A = adj^T tile row: two aligned 16B LDS loads
      const uint4 c0 = *(const uint4*)(&adjS[mloc * PADK + kh * 32 + 16 * hi]);
      const uint4 c1 = *(const uint4*)(&adjS[mloc * PADK + kh * 32 + 8 + 16 * hi]);
      const U32x8 ua = {c0, c1};
      const v16bf a = __builtin_bit_cast(v16bf, ua);

      // B tiles straight from xT (lane reads 16 consecutive n)
      const v16u b0u = *(const v16u*)(
          xT + ((size_t)b * HIDDEN + ht0 * 16 + l15) * NN + n0 + kh * 32 + 16 * hi);
      const v16u b1u = *(const v16u*)(
          xT + ((size_t)b * HIDDEN + (ht0 + 1) * 16 + l15) * NN + n0 + kh * 32 + 16 * hi);

      acc0 = __builtin_amdgcn_wmma_f32_16x16x32_bf16(
          false, a, false, __builtin_bit_cast(v16bf, b0u), (short)0, acc0, false, false);
      acc1 = __builtin_amdgcn_wmma_f32_16x16x32_bf16(
          false, a, false, __builtin_bit_cast(v16bf, b1u), (short)0, acc1, false, false);
    }
    __syncthreads();
  }

  // ---- transpose D tiles through LDS, write latT rows coalesced ----
#pragma unroll
  for (int r = 0; r < 8; ++r) {
    latS[(ht0 * 16 + l15) * 40 + msub * 16 + r + 8 * hi]       = f2bf(acc0[r]);
    latS[((ht0 + 1) * 16 + l15) * 40 + msub * 16 + r + 8 * hi] = f2bf(acc1[r]);
  }
  __syncthreads();
  const int h = tid >> 1, half = tid & 1;
  const uint4* src = (const uint4*)(&latS[h * 40 + half * 16]);
  uint4* dst = (uint4*)(latT + ((size_t)b * HIDDEN + h) * MM + m0 + half * 16);
  dst[0] = src[0];
  dst[1] = src[1];
}

// ---------------------------------------------------------------------------
// Stage 2: ret[b][n][h] = sum_m adj[b][n][m] * lat[b][m][h]
// adj tile is row-major in k already, so A is two aligned ds_load_b128.
// WG = 256 thr: 32(n) x 128(h) output tile, K-loop over M in steps of 128.
// ---------------------------------------------------------------------------
__global__ void __launch_bounds__(256)
stage2_out(const float* __restrict__ adj,
           const unsigned short* __restrict__ latT,
           float* __restrict__ out) {
  __shared__ unsigned short adjS[32 * PADK];  // bf16, [n_local][m_local]

  const int b   = blockIdx.y;
  const int n0  = blockIdx.x * 32;
  const int tid = threadIdx.x;
  const int wv = tid >> 5, lane = tid & 31;
  const int l15 = lane & 15, hi = lane >> 4;
  const int nsub = wv & 1;
  const int ht0  = (wv >> 1) * 2;
  const int nloc = nsub * 16 + l15;

  const int frow = tid >> 3;          // n_local 0..31
  const int fc   = (tid & 7) * 16;    // m_local base 0..112
  const float* fb = adj + ((size_t)b * NN + n0 + frow) * MM + fc;

  v8f acc0 = {}, acc1 = {};

  for (int kk = 0; kk < MM / 128; ++kk) {
    const int m0k = kk * 128;
    // ---- fill LDS: 64B per thread, packed bf16, 2x ds_store_b128 ----
    unsigned int dw[8];
#pragma unroll
    for (int j = 0; j < 4; ++j) {
      const float4 v = *(const float4*)(fb + m0k + 4 * j);
      dw[2 * j]     = (unsigned int)f2bf(v.x) | ((unsigned int)f2bf(v.y) << 16);
      dw[2 * j + 1] = (unsigned int)f2bf(v.z) | ((unsigned int)f2bf(v.w) << 16);
    }
    *(uint4*)(&adjS[frow * PADK + fc])     = make_uint4(dw[0], dw[1], dw[2], dw[3]);
    *(uint4*)(&adjS[frow * PADK + fc + 8]) = make_uint4(dw[4], dw[5], dw[6], dw[7]);
    __builtin_prefetch(fb + m0k + 128, 0, 1);
    __syncthreads();

#pragma unroll
    for (int kh = 0; kh < 4; ++kh) {
      const uint4 c0 = *(const uint4*)(&adjS[nloc * PADK + kh * 32 + 16 * hi]);
      const uint4 c1 = *(const uint4*)(&adjS[nloc * PADK + kh * 32 + 8 + 16 * hi]);
      const U32x8 ua = {c0, c1};
      const v16bf a = __builtin_bit_cast(v16bf, ua);

      const v16u b0u = *(const v16u*)(
          latT + ((size_t)b * HIDDEN + ht0 * 16 + l15) * MM + m0k + kh * 32 + 16 * hi);
      const v16u b1u = *(const v16u*)(
          latT + ((size_t)b * HIDDEN + (ht0 + 1) * 16 + l15) * MM + m0k + kh * 32 + 16 * hi);

      acc0 = __builtin_amdgcn_wmma_f32_16x16x32_bf16(
          false, a, false, __builtin_bit_cast(v16bf, b0u), (short)0, acc0, false, false);
      acc1 = __builtin_amdgcn_wmma_f32_16x16x32_bf16(
          false, a, false, __builtin_bit_cast(v16bf, b1u), (short)0, acc1, false, false);
    }
    __syncthreads();
  }

  // ---- store fp32 D tiles directly (row segments are 64B contiguous) ----
#pragma unroll
  for (int r = 0; r < 8; ++r) {
    const size_t row = (size_t)b * NN + n0 + nsub * 16 + r + 8 * hi;
    out[row * HIDDEN + ht0 * 16 + l15]       = acc0[r];
    out[row * HIDDEN + (ht0 + 1) * 16 + l15] = acc1[r];
  }
}

// ---------------------------------------------------------------------------
extern "C" void kernel_launch(void* const* d_in, const int* in_sizes, int n_in,
                              void* d_out, int out_size, void* d_ws, size_t ws_size,
                              hipStream_t stream) {
  (void)in_sizes; (void)n_in; (void)out_size; (void)ws_size;
  const float*         adj    = (const float*)d_in[0];
  const float*         embeds = (const float*)d_in[1];
  const int*           tindex = (const int*)d_in[2];
  const unsigned char* mask   = (const unsigned char*)d_in[3];
  const float*         W      = (const float*)d_in[4];
  const float*         bias   = (const float*)d_in[5];
  const float*         gamma  = (const float*)d_in[6];
  const float*         beta   = (const float*)d_in[7];

  unsigned short* xT   = (unsigned short*)d_ws;                    // [B][H][N] bf16 (8 MB)
  unsigned short* latT = xT + (size_t)BATCH * HIDDEN * NN;         // [B][H][M] bf16 (4 MB)
  float*          out  = (float*)d_out;

  stage0_linear_ln<<<BATCH * NN, HIDDEN, 0, stream>>>(
      embeds, tindex, mask, W, bias, gamma, beta, xT);
  stage1_aggT<<<dim3(MM / 32, BATCH), 256, 0, stream>>>(adj, xT, latT);
  stage2_out<<<dim3(NN / 32, BATCH), 256, 0, stream>>>(adj, latT, out);
}
